// Encoder_conv_mlp_86363202388083
// MI455X (gfx1250) — matrix operand order repaired and checked by reading, not compiled, not run on hardware
//
#include <hip/hip_runtime.h>
#include <hip/hip_bf16.h>

typedef __attribute__((ext_vector_type(16))) __bf16 v16bf;
typedef __attribute__((ext_vector_type(8)))  __bf16 v8bf;
typedef __attribute__((ext_vector_type(8)))  float  v8f;
typedef int v4i __attribute__((vector_size(16)));   // b128 payload type for async-LDS builtins

#define N_TOT   32768   // BS * N_NODES
#define N_NODES 64
#define BSZ     512
#define E_PER   1024
#define IN_F    128
#define HID     256
#define LAT     128

#if defined(__HIP_DEVICE_COMPILE__) && \
    __has_builtin(__builtin_amdgcn_global_load_async_to_lds_b128) && \
    __has_builtin(__builtin_amdgcn_s_wait_asynccnt)
#define HAVE_ASYNC_LDS 1
#define AS1P(p) ((__attribute__((address_space(1))) v4i*)(p))
#define AS3P(p) ((__attribute__((address_space(3))) v4i*)(p))
#endif

// ---------------- WMMA fragment loaders (row-major bf16, contiguous K) ----------------
// A (16x32): lane r (0-15) holds row M=r: K = k0+half*8 .. +7 in v[0..7], K+16 in v[8..15]
__device__ __forceinline__ v16bf load_frag_a(const __bf16* __restrict__ base, int row0,
                                             int ld, int k0, int lane) {
  const int half = lane >> 4, r = lane & 15;
  const __bf16* p = base + (size_t)(row0 + r) * ld + k0 + half * 8;
  v8bf lo = *(const v8bf*)(p);
  v8bf hi = *(const v8bf*)(p + 16);
  v16bf f;
#pragma unroll
  for (int i = 0; i < 8; ++i) { f[i] = lo[i]; f[i + 8] = hi[i]; }
  return f;
}

// B (32x16): lane r holds column N=r: K = k0+half*8 .. +7 in v[0..7], K+16 in v[8..15]
// weights are [N_out, K] row-major, so a column of B = one contiguous weight row.
__device__ __forceinline__ v16bf load_frag_b(const __bf16* __restrict__ base, int n0,
                                             int ld, int k0, int lane) {
  const int half = lane >> 4, r = lane & 15;
  const __bf16* p = base + (size_t)(n0 + r) * ld + k0 + half * 8;
  v8bf lo = *(const v8bf*)(p);
  v8bf hi = *(const v8bf*)(p + 16);
  v16bf f;
#pragma unroll
  for (int i = 0; i < 8; ++i) { f[i] = lo[i]; f[i + 8] = hi[i]; }
  return f;
}

// ---------------- f32 -> bf16 convert ----------------
extern "C" __global__ void cvt_bf16_kernel(const float* __restrict__ in,
                                           __bf16* __restrict__ out, int n) {
  int i = blockIdx.x * blockDim.x + threadIdx.x;
  int stride = gridDim.x * blockDim.x;
  for (; i < n; i += stride) out[i] = (__bf16)in[i];
}

// ---------------- per-graph segment-sum in LDS ----------------
// one block per graph: async-stage edges + 64xC activation tile straight into LDS
// (ASYNCcnt path on gfx1250), ds_add_f32 into an LDS accumulator, write bf16 result.
extern "C" __global__ __launch_bounds__(256)
void agg_kernel(const float* __restrict__ h, const int* __restrict__ ei,
                __bf16* __restrict__ aggb, int C, int logC, int E) {
  extern __shared__ float smem[];
  int*   es = (int*)smem;                 // [E_PER] src idx (raw, localized below)
  int*   ed = es + E_PER;                 // [E_PER] dst idx (contiguous with es)
  float* hs = (float*)(ed + E_PER);       // [64*C] staged activations
  float* ag = hs + N_NODES * C;           // [64*C] accumulator
  const int g = blockIdx.x, tid = threadIdx.x;
  const int nbase = g * N_NODES, ebase = g * E_PER;
  const int NC = N_NODES * C;

#ifdef HAVE_ASYNC_LDS
  {
    char* gs0 = (char*)(ei + ebase);          // src edge ids
    char* gs1 = (char*)(ei + E + ebase);      // dst edge ids
    for (int i = tid * 16; i < E_PER * 4; i += blockDim.x * 16) {
      __builtin_amdgcn_global_load_async_to_lds_b128(AS1P(gs0 + i), AS3P((char*)es + i), 0, 0);
      __builtin_amdgcn_global_load_async_to_lds_b128(AS1P(gs1 + i), AS3P((char*)ed + i), 0, 0);
    }
    char* gh = (char*)(h + (size_t)nbase * C);
    for (int i = tid * 16; i < NC * 4; i += blockDim.x * 16)
      __builtin_amdgcn_global_load_async_to_lds_b128(AS1P(gh + i), AS3P((char*)hs + i), 0, 0);
    for (int i = tid; i < NC; i += blockDim.x) ag[i] = 0.0f;  // overlaps async fill
    __builtin_amdgcn_s_wait_asynccnt(0);
    __syncthreads();
  }
#else
  for (int i = tid; i < E_PER; i += blockDim.x) {
    es[i] = ei[ebase + i];
    ed[i] = ei[E + ebase + i];
  }
  for (int i = tid; i < NC; i += blockDim.x) {
    hs[i] = h[(size_t)nbase * C + i];
    ag[i] = 0.0f;
  }
  __syncthreads();
#endif

  // localize global node ids -> graph-local (es/ed are contiguous)
  for (int i = tid; i < 2 * E_PER; i += blockDim.x) es[i] -= nbase;
  __syncthreads();

  const int maskC = C - 1;
  for (int i = tid; i < E_PER * C; i += blockDim.x) {
    int e = i >> logC, c = i & maskC;
    atomicAdd(&ag[ed[e] * C + c], hs[es[e] * C + c]);  // ds_add_f32
  }
  __syncthreads();
  for (int i = tid; i < NC; i += blockDim.x)
    aggb[(size_t)nbase * C + i] = (__bf16)ag[i];
}

// ---------------- fused two-term GEMM: relu(A0@B0^T + A1@B1^T + bias) ----------------
// block = 128 rows x 64 cols, 8 waves, each wave a 16x64 strip (4 wmma accumulators)
extern "C" __global__ __launch_bounds__(256)
void gemm_wmma(const __bf16* __restrict__ A0, const __bf16* __restrict__ B0,
               const __bf16* __restrict__ A1, const __bf16* __restrict__ B1,
               const float* __restrict__ bias,
               float* __restrict__ outF, __bf16* __restrict__ outB,
               int N, int K, int doRelu) {
  const int lane = threadIdx.x & 31;
  const int wave = threadIdx.x >> 5;
  const int row0 = blockIdx.x * 128 + wave * 16;
  const int n0   = blockIdx.y * 64;
  v8f acc[4] = {};
  for (int t = 0; t < 2; ++t) {
    const __bf16* A = t ? A1 : A0;
    const __bf16* B = t ? B1 : B0;
    if (!A) break;
    for (int k0 = 0; k0 < K; k0 += 32) {
      v16bf a = load_frag_a(A, row0, K, k0, lane);
#pragma unroll
      for (int j = 0; j < 4; ++j) {
        v16bf b = load_frag_b(B, n0 + j * 16, K, k0, lane);
        acc[j] = __builtin_amdgcn_wmma_f32_16x16x32_bf16(
            false, a, false, b, (short)0, acc[j], false, false);
      }
    }
  }
  const int half = lane >> 4, nl = lane & 15;
#pragma unroll
  for (int j = 0; j < 4; ++j) {
    const int col = n0 + j * 16 + nl;
    const float bv = bias ? bias[col] : 0.0f;
#pragma unroll
    for (int i = 0; i < 8; ++i) {
      const int row = row0 + half * 8 + i;
      float v = acc[j][i] + bv;
      if (doRelu) v = fmaxf(v, 0.0f);
      const size_t idx = (size_t)row * N + col;
      if (outF) outF[idx] = v;
      if (outB) outB[idx] = (__bf16)v;
    }
  }
}

// ---------------- head GEMM with split-K + f32 atomic reduction ----------------
extern "C" __global__ __launch_bounds__(256)
void gemm_head(const __bf16* __restrict__ A, const __bf16* __restrict__ B,
               float* __restrict__ out, int N, int K, int kChunk) {
  const int lane = threadIdx.x & 31;
  const int wave = threadIdx.x >> 5;
  const int row0 = blockIdx.x * 128 + wave * 16;
  const int n0   = blockIdx.y * 64;
  const int kb   = blockIdx.z * kChunk;
  v8f acc[4] = {};
  for (int k0 = kb; k0 < kb + kChunk; k0 += 32) {
    v16bf a = load_frag_a(A, row0, K, k0, lane);
#pragma unroll
    for (int j = 0; j < 4; ++j) {
      v16bf b = load_frag_b(B, n0 + j * 16, K, k0, lane);
      acc[j] = __builtin_amdgcn_wmma_f32_16x16x32_bf16(
          false, a, false, b, (short)0, acc[j], false, false);
    }
  }
  const int half = lane >> 4, nl = lane & 15;
#pragma unroll
  for (int j = 0; j < 4; ++j) {
    const int col = n0 + j * 16 + nl;
#pragma unroll
    for (int i = 0; i < 8; ++i) {
      const int row = row0 + half * 8 + i;
      atomicAdd(&out[(size_t)row * N + col], acc[j][i]);
    }
  }
}

// ---------------- init d_out with biases (atomics accumulate on top) ----------------
extern "C" __global__ void init_out_kernel(float* __restrict__ out,
                                           const float* __restrict__ bmu,
                                           const float* __restrict__ blv) {
  int i = blockIdx.x * blockDim.x + threadIdx.x;
  if (i < BSZ * LAT) {
    out[i] = bmu[i & (LAT - 1)];
    out[BSZ * LAT + i] = blv[i & (LAT - 1)];
  }
}

extern "C" void kernel_launch(void* const* d_in, const int* in_sizes, int n_in,
                              void* d_out, int out_size, void* d_ws, size_t ws_size,
                              hipStream_t stream) {
  const float* x       = (const float*)d_in[0];
  const float* W1_rel  = (const float*)d_in[1];
  const float* W1_root = (const float*)d_in[2];
  const float* b1      = (const float*)d_in[3];
  const float* W2_rel  = (const float*)d_in[4];
  const float* W2_root = (const float*)d_in[5];
  const float* b2      = (const float*)d_in[6];
  const float* Wmu     = (const float*)d_in[7];
  const float* bmu     = (const float*)d_in[8];
  const float* Wlv     = (const float*)d_in[9];
  const float* blv     = (const float*)d_in[10];
  const int*   ei      = (const int*)d_in[11];
  const int E = in_sizes[11] / 2;

  // workspace carve-up (256B aligned)
  size_t off = 0;
  char* base = (char*)d_ws;
  auto wsalloc = [&](size_t bytes) -> void* {
    void* p = base + off;
    off += (bytes + 255) & ~(size_t)255;
    return p;
  };
  __bf16* xb      = (__bf16*)wsalloc((size_t)N_TOT * IN_F * 2);
  __bf16* w1relb  = (__bf16*)wsalloc((size_t)HID * IN_F * 2);
  __bf16* w1rootb = (__bf16*)wsalloc((size_t)HID * IN_F * 2);
  __bf16* w2relb  = (__bf16*)wsalloc((size_t)HID * HID * 2);
  __bf16* w2rootb = (__bf16*)wsalloc((size_t)HID * HID * 2);
  __bf16* wmub    = (__bf16*)wsalloc((size_t)LAT * N_NODES * HID * 2);
  __bf16* wlvb    = (__bf16*)wsalloc((size_t)LAT * N_NODES * HID * 2);
  __bf16* agg1b   = (__bf16*)wsalloc((size_t)N_TOT * IN_F * 2);
  float*  h1f     = (float*) wsalloc((size_t)N_TOT * HID * 4);
  __bf16* h1b     = (__bf16*)wsalloc((size_t)N_TOT * HID * 2);
  __bf16* agg2b   = (__bf16*)wsalloc((size_t)N_TOT * HID * 2);
  __bf16* h2b     = (__bf16*)wsalloc((size_t)N_TOT * HID * 2);

  auto cvt = [&](const float* in, __bf16* out, size_t n) {
    int blocks = (int)((n + 255) / 256);
    if (blocks > 4096) blocks = 4096;
    cvt_bf16_kernel<<<blocks, 256, 0, stream>>>(in, out, (int)n);
  };
  cvt(x,       xb,      (size_t)N_TOT * IN_F);
  cvt(W1_rel,  w1relb,  (size_t)HID * IN_F);
  cvt(W1_root, w1rootb, (size_t)HID * IN_F);
  cvt(W2_rel,  w2relb,  (size_t)HID * HID);
  cvt(W2_root, w2rootb, (size_t)HID * HID);
  cvt(Wmu,     wmub,    (size_t)LAT * N_NODES * HID);
  cvt(Wlv,     wlvb,    (size_t)LAT * N_NODES * HID);

  // layer 1: agg1 = segment_sum(x[src], dst); h1 = relu(agg1@W1_rel^T + x@W1_root^T + b1)
  size_t smem1 = 2 * E_PER * 4 + 2 * (size_t)N_NODES * IN_F * 4;   // 73,728 B
  agg_kernel<<<BSZ, 256, smem1, stream>>>(x, ei, agg1b, IN_F, 7, E);
  gemm_wmma<<<dim3(N_TOT / 128, HID / 64), 256, 0, stream>>>(
      agg1b, w1relb, xb, w1rootb, b1, h1f, h1b, HID, IN_F, 1);

  // layer 2: agg2 = segment_sum(h1[src], dst); h2 = relu(agg2@W2_rel^T + h1@W2_root^T + b2)
  size_t smem2 = 2 * E_PER * 4 + 2 * (size_t)N_NODES * HID * 4;    // 139,264 B (<320KB WGP LDS)
  agg_kernel<<<BSZ, 256, smem2, stream>>>(h1f, ei, agg2b, HID, 8, E);
  gemm_wmma<<<dim3(N_TOT / 128, HID / 64), 256, 0, stream>>>(
      agg2b, w2relb, h1b, w2rootb, b2, nullptr, h2b, HID, HID, 1);

  // heads: hb = h2 viewed as [512, 16384]; mu/logvar via split-K + atomic f32 reduce
  float* out_mu = (float*)d_out;
  float* out_lv = out_mu + (size_t)BSZ * LAT;
  init_out_kernel<<<(BSZ * LAT + 255) / 256, 256, 0, stream>>>(out_mu, bmu, blv);
  const int Khead = N_NODES * HID;          // 16384
  const int kChunk = Khead / 8;             // 2048
  gemm_head<<<dim3(BSZ / 128, LAT / 64, 8), 256, 0, stream>>>(h2b, wmub, out_mu, LAT, Khead, kChunk);
  gemm_head<<<dim3(BSZ / 128, LAT / 64, 8), 256, 0, stream>>>(h2b, wlvb, out_lv, LAT, Khead, kChunk);
}
